// MegatronAttention_69432441307486
// MI455X (gfx1250) — matrix operand order repaired and checked by reading, not compiled, not run on hardware
//
#include <hip/hip_runtime.h>
#include <stdint.h>

// ---------------------------------------------------------------------------
// Problem constants (from reference)
// ---------------------------------------------------------------------------
constexpr int cB   = 2;
constexpr int cS   = 2048;
constexpr int cHID = 3072;
constexpr int cNH  = 24;
constexpr int cNKV = 8;
constexpr int cHD  = 128;
constexpr int cG   = cNH / cNKV;          // 3
constexpr int cM   = cB * cS;             // 4096 rows
constexpr int cKV  = cNKV * cHD;          // 1024

// ---------------------------------------------------------------------------
// WMMA helpers (CDNA5: v_wmma_f32_16x16x32_bf16, wave32)
// ---------------------------------------------------------------------------
typedef __bf16 v16bf __attribute__((ext_vector_type(16)));
typedef float  v8f   __attribute__((ext_vector_type(8)));

union FragU { v16bf v; uint4 q[2]; };

__device__ __forceinline__ v16bf load_frag(const unsigned short* p0,
                                           const unsigned short* p1) {
  FragU f;
  f.q[0] = *(const uint4*)p0;
  f.q[1] = *(const uint4*)p1;
  return f.v;
}

__device__ __forceinline__ v8f wmma_bf16(v16bf a, v16bf b, v8f c) {
  // 8 args: (neg_a, A, neg_b, B, c_mod, C, reuse_a, reuse_b)
  return __builtin_amdgcn_wmma_f32_16x16x32_bf16(false, a, false, b,
                                                 (short)0, c, false, false);
}

__device__ __forceinline__ unsigned short f2bf(float x) {
  unsigned int u = __float_as_uint(x);
  unsigned int r = u + 0x7FFFu + ((u >> 16) & 1u);
  return (unsigned short)(r >> 16);
}

__device__ __forceinline__ v8f zero8() {
  v8f z;
#pragma unroll
  for (int i = 0; i < 8; ++i) z[i] = 0.0f;
  return z;
}

// ---------------------------------------------------------------------------
// fp32 -> bf16 conversion
// ---------------------------------------------------------------------------
__global__ void cvt_bf16_kernel(const float* __restrict__ in,
                                unsigned short* __restrict__ out, int n) {
  int i = blockIdx.x * blockDim.x + threadIdx.x;
  int stride = gridDim.x * blockDim.x;
  for (; i < n; i += stride) out[i] = f2bf(in[i]);
}

// ---------------------------------------------------------------------------
// Tiled bf16 GEMM:  out[M,N] = A[M,K] * W[N,K]^T + bias[N]
// block = 128 threads (4 waves, 2x2 grid); block tile 128x128; K stepped by
// 64 with double-buffered LDS. Each wave computes a 64x64 sub-tile:
// 16 accumulators, 16 WMMAs per 32-K step with only 8 fragment loads
// (1.0 ds_load_b128 per WMMA). Next tile's global loads are split into two
// halves (A, then W) and issued between the compute phases for overlap.
// ---------------------------------------------------------------------------
template <bool OUT_BF16>
__global__ __launch_bounds__(128) void gemm_bf16_kernel(
    const unsigned short* __restrict__ A, const unsigned short* __restrict__ W,
    const float* __restrict__ bias, void* __restrict__ out, int M, int N, int K) {
  constexpr int LDT = 72;  // padded LDS row stride (shorts); 144B, 16B aligned
  __shared__ unsigned short As[2][128 * LDT];  // 2 x 18,432 B
  __shared__ unsigned short Ws[2][128 * LDT];  // 2 x 18,432 B

  const int t    = threadIdx.x;
  const int lane = t & 31;
  const int wave = t >> 5;
  const int hl   = lane >> 4;   // lane half (0/1)
  const int l15  = lane & 15;
  const int wr   = wave >> 1;   // wave row in 2x2 grid
  const int wc   = wave & 1;    // wave col in 2x2 grid
  const int m0   = blockIdx.x * 128;
  const int n0   = blockIdx.y * 128;

  const int akb = hl ? 8 : 0;    // A-frag K sub-base within 32-chunk
  const int bkb = hl ? 8 : 0;    // B-frag K sub-base within 32-chunk

  v8f acc[4][4];
#pragma unroll
  for (int i = 0; i < 4; ++i)
#pragma unroll
    for (int j = 0; j < 4; ++j) acc[i][j] = zero8();

  uint4 ra[8], rw[8];  // staging registers (global -> reg -> LDS)

  auto gloadA = [&](int k0) {
#pragma unroll
    for (int i = 0; i < 8; ++i) {  // 128 rows x 128B = 1024 uint4
      int idx = i * 128 + t;
      int r = idx >> 3, c = idx & 7;
      ra[i] = *(const uint4*)&A[(size_t)(m0 + r) * K + k0 + c * 8];
    }
  };
  auto gloadW = [&](int k0) {
#pragma unroll
    for (int i = 0; i < 8; ++i) {
      int idx = i * 128 + t;
      int r = idx >> 3, c = idx & 7;
      rw[i] = *(const uint4*)&W[(size_t)(n0 + r) * K + k0 + c * 8];
    }
  };
  auto lstore = [&](int buf) {
#pragma unroll
    for (int i = 0; i < 8; ++i) {
      int idx = i * 128 + t;
      int r = idx >> 3, c = idx & 7;
      *(uint4*)&As[buf][r * LDT + c * 8] = ra[i];
      *(uint4*)&Ws[buf][r * LDT + c * 8] = rw[i];
    }
  };

  // One 32-K compute step on LDS buffer `cur`, K sub-offset kb (0 or 32).
  auto compute = [&](int cur, int kb) {
    v16bf af[4];
#pragma unroll
    for (int i = 0; i < 4; ++i) {
      const unsigned short* ap =
          &As[cur][(wr * 64 + i * 16 + l15) * LDT + kb + akb];
      af[i] = load_frag(ap, ap + 16);
    }
#pragma unroll
    for (int nf = 0; nf < 4; ++nf) {
      const unsigned short* bp =
          &Ws[cur][(wc * 64 + nf * 16 + l15) * LDT + kb + bkb];
      v16bf b = load_frag(bp, bp + 16);
#pragma unroll
      for (int i = 0; i < 4; ++i) acc[i][nf] = wmma_bf16(af[i], b, acc[i][nf]);
    }
  };

  gloadA(0);
  gloadW(0);
  lstore(0);
  __syncthreads();

  const int KT = K >> 6;  // K / 64
  for (int kt = 0; kt < KT; ++kt) {
    const int cur = kt & 1;
    const bool more = (kt + 1 < KT);
    if (more) gloadA((kt + 1) * 64);  // overlaps with ks=0 WMMAs
    compute(cur, 0);
    if (more) gloadW((kt + 1) * 64);  // overlaps with ks=1 WMMAs
    compute(cur, 32);
    if (more) {
      __syncthreads();          // everyone done reading buffer cur^1
      lstore(cur ^ 1);
      __syncthreads();          // next buffer visible
    }
  }

  // Epilogue: C layout -> lane l15 = col, VGPR r = row (r or r+8 by lane half)
#pragma unroll
  for (int af = 0; af < 4; ++af) {
#pragma unroll
    for (int nf = 0; nf < 4; ++nf) {
      int n = n0 + wc * 64 + nf * 16 + l15;
      float bv = bias[n];
#pragma unroll
      for (int r = 0; r < 8; ++r) {
        int m = m0 + wr * 64 + af * 16 + r + 8 * hl;
        float val = acc[af][nf][r] + bv;
        if (OUT_BF16)
          ((unsigned short*)out)[(size_t)m * N + n] = f2bf(val);
        else
          ((float*)out)[(size_t)m * N + n] = val;
      }
    }
  }
}

// ---------------------------------------------------------------------------
// Flash-style causal GQA attention.
//  Q: [B,S,NH,HD] bf16   K,V: [B,S,NKV,HD] bf16   O: [B,S,NH*HD] bf16
// grid = (S/64, B*NH); block = 128 (4 waves); wave owns 16 q rows.
// Double-buffered V^T tile -> one barrier per k-tile; K fragments are
// software-pipelined from global (load dk+1 while computing dk).
// ---------------------------------------------------------------------------
__global__ __launch_bounds__(128) void attn_kernel(
    const unsigned short* __restrict__ Q, const unsigned short* __restrict__ Kc,
    const unsigned short* __restrict__ Vc, unsigned short* __restrict__ O) {
  constexpr int LV = 72;  // V^T LDS row stride (shorts), 144B (16B aligned)
  constexpr int LP = 72;  // P  LDS row stride
  __shared__ unsigned short Vt[2][cHD * LV];  // V transposed: [d][kcol], 2x18.4KB
  __shared__ unsigned short Ps[64 * LP];      // probabilities (wave-private rows)

  const int t    = threadIdx.x;
  const int lane = t & 31;
  const int wave = t >> 5;
  const int hl   = lane >> 4;
  const int l15  = lane & 15;
  const int qt   = blockIdx.x;              // q tile (64 rows)
  const int bh   = blockIdx.y;              // b * NH + h
  const int b    = bh / cNH;
  const int h    = bh % cNH;
  const int kv   = h / cG;
  const float scale = 0.08838834764831845f;  // HD^-0.5

  const unsigned short* qb = Q  + ((size_t)(b * cS) * cNH  + h)  * cHD;
  const unsigned short* kb = Kc + ((size_t)(b * cS) * cNKV + kv) * cHD;
  const unsigned short* vb = Vc + ((size_t)(b * cS) * cNKV + kv) * cHD;

  const int akb = hl ? 8 : 0;
  const int bkb = hl ? 8 : 0;

  // Stage one 64x128 V tile transposed into LDS buffer `buf`.
  auto stageV = [&](int kt, int buf) {
#pragma unroll
    for (int i = 0; i < 8; ++i) {
      int idx  = i * 128 + t;
      int kcol = idx >> 4;         // 0..63
      int d0   = (idx & 15) * 8;   // 0..120
      uint4 vv = *(const uint4*)&vb[(size_t)(kt * 64 + kcol) * cNKV * cHD + d0];
      const unsigned short* e = (const unsigned short*)&vv;
#pragma unroll
      for (int j = 0; j < 8; ++j) Vt[buf][(d0 + j) * LV + kcol] = e[j];
    }
  };

  // Q fragments (A operand): 4 chunks of 32 along D, resident in registers.
  v16bf qf[4];
  {
    int srow = qt * 64 + wave * 16 + l15;
    const unsigned short* qr = qb + (size_t)srow * cNH * cHD;
#pragma unroll
    for (int dk = 0; dk < 4; ++dk) {
      const unsigned short* p = qr + dk * 32 + akb;
      qf[dk] = load_frag(p, p + 16);
    }
  }

  v8f ctx[8];
  float m_i[8], l_i[8];
#pragma unroll
  for (int i = 0; i < 8; ++i) { ctx[i] = zero8(); m_i[i] = -3.0e38f; l_i[i] = 0.0f; }

  stageV(0, 0);
  __syncthreads();

  for (int kt = 0; kt <= qt; ++kt) {
    const int cur = kt & 1;
    if (kt < qt) stageV(kt + 1, cur ^ 1);  // overlaps with compute below

    // --- scores: S = Q * K^T, K fragments pipelined from global ---
    v8f sa[4];
#pragma unroll
    for (int i = 0; i < 4; ++i) sa[i] = zero8();

    const unsigned short* kr[4];
#pragma unroll
    for (int nf = 0; nf < 4; ++nf)
      kr[nf] = kb + (size_t)(kt * 64 + nf * 16 + l15) * cNKV * cHD + bkb;

    v16bf kf[4];
#pragma unroll
    for (int nf = 0; nf < 4; ++nf) kf[nf] = load_frag(kr[nf], kr[nf] + 16);

#pragma unroll
    for (int dk = 0; dk < 4; ++dk) {
      v16bf kn[4];
      if (dk < 3) {
#pragma unroll
        for (int nf = 0; nf < 4; ++nf) {
          const unsigned short* p = kr[nf] + (dk + 1) * 32;
          kn[nf] = load_frag(p, p + 16);
        }
      }
#pragma unroll
      for (int nf = 0; nf < 4; ++nf) sa[nf] = wmma_bf16(qf[dk], kf[nf], sa[nf]);
      if (dk < 3) {
#pragma unroll
        for (int nf = 0; nf < 4; ++nf) kf[nf] = kn[nf];
      }
    }

    // --- online softmax (rows in VGPR index, cols across 16-lane halves) ---
    const bool diag = (kt == qt);  // uniform: only diagonal tile needs masking
#pragma unroll
    for (int r = 0; r < 8; ++r) {
      float sv[4];
      float vmax = -3.0e38f;
#pragma unroll
      for (int nf = 0; nf < 4; ++nf) {
        float x = sa[nf][r] * scale;
        if (diag) {
          int rowg = qt * 64 + wave * 16 + r + 8 * hl;
          int colg = kt * 64 + nf * 16 + l15;
          if (colg > rowg) x = -1.0e30f;  // causal mask
        }
        sv[nf] = x;
        vmax = fmaxf(vmax, x);
      }
#pragma unroll
      for (int m = 1; m < 16; m <<= 1) vmax = fmaxf(vmax, __shfl_xor(vmax, m, 32));
      float mnew = fmaxf(m_i[r], vmax);
      float corr = __expf(m_i[r] - mnew);
      float rs = 0.0f;
#pragma unroll
      for (int nf = 0; nf < 4; ++nf) {
        float p = __expf(sv[nf] - mnew);
        sv[nf] = p;
        rs += p;
      }
#pragma unroll
      for (int m = 1; m < 16; m <<= 1) rs += __shfl_xor(rs, m, 32);
      l_i[r] = l_i[r] * corr + rs;
      m_i[r] = mnew;
#pragma unroll
      for (int nf2 = 0; nf2 < 8; ++nf2) ctx[nf2][r] *= corr;
      int prow = wave * 16 + r + 8 * hl;
#pragma unroll
      for (int nf = 0; nf < 4; ++nf)
        Ps[prow * LP + nf * 16 + l15] = f2bf(sv[nf]);
    }
    // Ps rows are wave-private; same-wave LDS ops are in-order -> no barrier.

    // --- ctx += P * V (A from Ps, B from transposed Vt[cur]) ---
#pragma unroll
    for (int kk = 0; kk < 2; ++kk) {
      const unsigned short* pp = &Ps[(wave * 16 + l15) * LP + kk * 32 + akb];
      v16bf pf = load_frag(pp, pp + 16);
#pragma unroll
      for (int nf2 = 0; nf2 < 8; ++nf2) {
        const unsigned short* vp = &Vt[cur][(nf2 * 16 + l15) * LV + kk * 32 + bkb];
        v16bf vf = load_frag(vp, vp + 16);
        ctx[nf2] = wmma_bf16(pf, vf, ctx[nf2]);
      }
    }

    __syncthreads();  // Vt[cur] reads done; Vt[cur^1] writes visible
  }

  // --- finalize: divide by row sum, store bf16 ctx [B,S,NH*HD] ---
#pragma unroll
  for (int r = 0; r < 8; ++r) {
    float inv = 1.0f / l_i[r];
    int srow = qt * 64 + wave * 16 + r + 8 * hl;
    unsigned short* orow = O + ((size_t)(b * cS + srow) * cNH + h) * cHD;
#pragma unroll
    for (int nf2 = 0; nf2 < 8; ++nf2)
      orow[nf2 * 16 + l15] = f2bf(ctx[nf2][r] * inv);
  }
}

// ---------------------------------------------------------------------------
// Host-side orchestration
// ---------------------------------------------------------------------------
static inline void cvt(const float* in, unsigned short* out, int n, hipStream_t s) {
  int blocks = (n + 256 * 8 - 1) / (256 * 8);
  if (blocks > 4096) blocks = 4096;
  cvt_bf16_kernel<<<blocks, 256, 0, s>>>(in, out, n);
}

extern "C" void kernel_launch(void* const* d_in, const int* in_sizes, int n_in,
                              void* d_out, int out_size, void* d_ws, size_t ws_size,
                              hipStream_t stream) {
  (void)in_sizes; (void)n_in; (void)out_size; (void)ws_size;
  const float* x  = (const float*)d_in[0];
  // d_in[1] = attention_mask (pure causal -> computed inline)
  const float* Wq = (const float*)d_in[2];
  const float* bq = (const float*)d_in[3];
  const float* Wk = (const float*)d_in[4];
  const float* bk = (const float*)d_in[5];
  const float* Wv = (const float*)d_in[6];
  const float* bv = (const float*)d_in[7];
  const float* Wd = (const float*)d_in[8];
  const float* bd = (const float*)d_in[9];
  float* out = (float*)d_out;

  char* ws = (char*)d_ws;
  size_t off = 0;
  auto alloc = [&](size_t elems) {
    unsigned short* p = (unsigned short*)(ws + off);
    off += ((elems * 2 + 255) / 256) * 256;
    return p;
  };
  unsigned short* Xb  = alloc((size_t)cM * cHID);
  unsigned short* Wqb = alloc((size_t)cHID * cHID);
  unsigned short* Wkb = alloc((size_t)cKV * cHID);
  unsigned short* Wvb = alloc((size_t)cKV * cHID);
  unsigned short* Wdb = alloc((size_t)cHID * cHID);
  unsigned short* Qb  = alloc((size_t)cM * cHID);
  unsigned short* Kb  = alloc((size_t)cM * cKV);
  unsigned short* Vb  = alloc((size_t)cM * cKV);
  unsigned short* Cb  = alloc((size_t)cM * cHID);

  // 1) fp32 -> bf16
  cvt(x,  Xb,  cM * cHID,   stream);
  cvt(Wq, Wqb, cHID * cHID, stream);
  cvt(Wk, Wkb, cKV * cHID,  stream);
  cvt(Wv, Wvb, cKV * cHID,  stream);
  cvt(Wd, Wdb, cHID * cHID, stream);

  // 2) QKV projections (bf16 out)
  gemm_bf16_kernel<true><<<dim3(cM / 128, cHID / 128), 128, 0, stream>>>(
      Xb, Wqb, bq, Qb, cM, cHID, cHID);
  gemm_bf16_kernel<true><<<dim3(cM / 128, cKV / 128), 128, 0, stream>>>(
      Xb, Wkb, bk, Kb, cM, cKV, cHID);
  gemm_bf16_kernel<true><<<dim3(cM / 128, cKV / 128), 128, 0, stream>>>(
      Xb, Wvb, bv, Vb, cM, cKV, cHID);

  // 3) causal GQA flash attention
  attn_kernel<<<dim3(cS / 64, cB * cNH), 128, 0, stream>>>(Qb, Kb, Vb, Cb);

  // 4) output projection (fp32 out)
  gemm_bf16_kernel<false><<<dim3(cM / 128, cHID / 128), 128, 0, stream>>>(
      Cb, Wdb, bd, out, cM, cHID, cHID);
}
